// MyRNN_12549894438919
// MI455X (gfx1250) — matrix operand order repaired and checked by reading, not compiled, not run on hardware
//
#include <hip/hip_runtime.h>
#include <math.h>

// ---------------------------------------------------------------------------
// Multi-layer tanh RNN for MI455X (gfx1250, wave32, WMMA, async-to-LDS).
//   h_t = tanh(x_t @ W_ih^T + h_{t-1} @ W_hh^T)
// One-time fp32->bf16 conversion kernels feed:
//   Phase 1 per layer: 128x64-tile bf16 WMMA GEMM (input projection).
//   Phase 2 per layer: persistent scan, W_hh slice pinned in LDS, bf16 hidden
//                      state ping-pong, device-scope barrier per step.
// ---------------------------------------------------------------------------

#define USE_ASYNC_COPY 1

#define SEQLEN 2048
#define BATCH  32
#define HDIM   1024
#define NWG    32                 // workgroups in the persistent scan
#define NSLICE (HDIM / NWG)       // 32 output columns per workgroup
#define GT_M   128                // GEMM block tile M
#define GT_N   64                 // GEMM block tile N
#define WPAD   40                 // LDS row pad: 80B stride, 16B aligned
#define HPAD   1040               // 2080B stride, 16B aligned

typedef __attribute__((ext_vector_type(16))) __bf16 v16bf;
typedef __attribute__((ext_vector_type(8)))  float  v8f;

struct alignas(16) U4 { unsigned v[4]; };          // trivially-constructible 16B

union FragBF {
  v16bf v;
  unsigned short u[16];
  U4 q[2];
};

__device__ __forceinline__ unsigned short f2bf(float f) {
  // round-to-nearest-even fp32 -> bf16
  unsigned u = __float_as_uint(f);
  u += 0x7FFFu + ((u >> 16) & 1u);
  return (unsigned short)(u >> 16);
}

__device__ __forceinline__ unsigned lds_off(const void* p) {
  // flat LDS aperture: addr[31:0] is the wave-relative LDS byte offset
  return (unsigned)(unsigned long long)p;
}

__device__ __forceinline__ void async_g2l_b128(unsigned lds_byte_off,
                                               const void* gaddr) {
  // GLOBAL_LOAD_ASYNC_TO_LDS_B128: per-lane 16B gather into LDS (ASYNCcnt)
  asm volatile("global_load_async_to_lds_b128 %0, %1, off"
               :: "v"(lds_byte_off), "v"((unsigned long long)gaddr)
               : "memory");
}

__device__ __forceinline__ void wait_async0() {
  asm volatile("s_wait_asynccnt 0" ::: "memory");
}

// ---------------------------------------------------------------------------
// One-time converters (memory-bound, ~20us total at 23.3 TB/s).
// ---------------------------------------------------------------------------
__global__ __launch_bounds__(256) void cvt_bf16(const float* __restrict__ s,
                                                unsigned short* __restrict__ d,
                                                size_t n8) {   // count/8
  size_t i = (size_t)blockIdx.x * 256 + threadIdx.x;
  if (i >= n8) return;
  const float* p = s + i * 8;
  U4 o;
#pragma unroll
  for (int j = 0; j < 4; ++j)
    o.v[j] = (unsigned)f2bf(p[2 * j]) | ((unsigned)f2bf(p[2 * j + 1]) << 16);
  *(U4*)(d + i * 8) = o;
}

// W [N,K] fp32 -> Wt [K,N] bf16 (pre-transposed so B-fragments are contiguous)
__global__ __launch_bounds__(256) void tcvt_bf16(const float* __restrict__ W,
                                                 unsigned short* __restrict__ Wt) {
  int idx = blockIdx.x * 256 + threadIdx.x;       // 0 .. HDIM*HDIM-1
  int k = idx >> 10, n = idx & (HDIM - 1);
  Wt[(size_t)k * HDIM + n] = f2bf(W[(size_t)n * HDIM + k]);
}

// ---------------------------------------------------------------------------
// C[M,N] = A[M,K](bf16) * Wt[K,N](bf16), fp32 accumulate.
// Block 128x64, 8 waves, each wave does a 32x32 region = 2x2 WMMA tiles.
// A panel staged through LDS (async); B read straight from L2-resident Wt.
// ---------------------------------------------------------------------------
__global__ __launch_bounds__(256) void gemm_bf16(const unsigned short* __restrict__ A,
                                                 const unsigned short* __restrict__ Wt,
                                                 float* __restrict__ C) {
  __shared__ unsigned short lA[GT_M][WPAD];

  const int tid  = threadIdx.x;
  const int lane = tid & 31;
  const int wv   = tid >> 5;
  const int mp   = wv & 3;           // 32-row group
  const int np   = wv >> 2;          // 32-col group
  const size_t m0 = (size_t)blockIdx.y * GT_M;
  const int    n0 = blockIdx.x * GT_N;

  v8f acc[2][2] = {};
  for (int kt = 0; kt < HDIM; kt += 32) {
    // stage A tile 128x32 bf16 (8KB): 512 x 16B chunks, 2 per thread
#pragma unroll
    for (int i = 0; i < 2; ++i) {
      int c = i * 256 + tid;
      int r = c >> 2, p = c & 3;
#if USE_ASYNC_COPY
      async_g2l_b128(lds_off(&lA[r][p * 8]),
                     A + (m0 + r) * HDIM + kt + p * 8);
#else
      *(U4*)&lA[r][p * 8] = *(const U4*)(A + (m0 + r) * HDIM + kt + p * 8);
#endif
    }
#if USE_ASYNC_COPY
    wait_async0();
#endif
    __syncthreads();

    FragBF fa[2], fb[2];
    {
      int rbase = mp * 32 + (lane & 15);
      int kb    = (lane >> 4) * 8;
#pragma unroll
      for (int a = 0; a < 2; ++a) {
        int row = rbase + a * 16;
#pragma unroll
        for (int i = 0; i < 8; ++i) {
          fa[a].u[i]     = lA[row][kb + i];
          fa[a].u[8 + i] = lA[row][kb + 16 + i];
        }
      }
      // B fragment: lane holds column N=lane&15, K in table order (matches A)
      const unsigned short* wcol = Wt + (size_t)(kt + kb) * HDIM + n0 + np * 32 + (lane & 15);
#pragma unroll
      for (int b = 0; b < 2; ++b) {
#pragma unroll
        for (int i = 0; i < 8; ++i) {
          fb[b].u[i]     = wcol[(size_t)i * HDIM + b * 16];
          fb[b].u[8 + i] = wcol[(size_t)(16 + i) * HDIM + b * 16];
        }
      }
    }
#pragma unroll
    for (int a = 0; a < 2; ++a)
#pragma unroll
      for (int b = 0; b < 2; ++b)
        acc[a][b] = __builtin_amdgcn_wmma_f32_16x16x32_bf16(
            false, fa[a].v, false, fb[b].v, (short)0, acc[a][b], false, false);
    __syncthreads();
  }

  // C/D layout: VGPR r -> M=r (lanes 0-15) / M=r+8 (lanes 16-31), N=lane&15
#pragma unroll
  for (int a = 0; a < 2; ++a) {
    size_t mbase = m0 + mp * 32 + a * 16 + ((lane >> 4) << 3);
#pragma unroll
    for (int b = 0; b < 2; ++b) {
      int n = n0 + np * 32 + b * 16 + (lane & 15);
#pragma unroll
      for (int r = 0; r < 8; ++r)
        C[(mbase + r) * HDIM + n] = acc[a][b][r];
    }
  }
}

// ---------------------------------------------------------------------------
// Persistent recurrent scan. NWG blocks x 128 threads (4 waves).
// W_hh^T slice pinned in LDS (bf16) for all 2048 steps; hidden state
// ping-pongs through global as bf16 and is async-copied into LDS per step.
// ---------------------------------------------------------------------------
__global__ __launch_bounds__(128) void rnn_scan(const float* __restrict__ xp,  // [S,B,H] fp32
                                                float* __restrict__ outF,      // [S,B,H] or null
                                                unsigned short* __restrict__ outBF, // [S,B,H] or null
                                                const unsigned short* __restrict__ WtHH, // [K,N] bf16
                                                unsigned short* __restrict__ Ha,
                                                unsigned short* __restrict__ Hb,
                                                float* __restrict__ hn,        // [B,H]
                                                unsigned* __restrict__ bar) {
  extern __shared__ unsigned short smem[];
  unsigned short (*lW)[WPAD] = (unsigned short (*)[WPAD])smem;                 // [HDIM][WPAD]
  unsigned short (*lH)[HPAD] = (unsigned short (*)[HPAD])(smem + HDIM * WPAD); // [BATCH][HPAD]

  const int tid  = threadIdx.x;
  const int lane = tid & 31;
  const int wv   = tid >> 5;
  const int mi   = wv & 1;           // batch-row tile 0..1
  const int ni   = wv >> 1;          // N tile 0..1 within slice
  const int n0   = blockIdx.x * NSLICE;

  // Pin W_hh^T slice [HDIM][NSLICE] in LDS — once. 4096 x 16B chunks.
#pragma unroll
  for (int i = 0; i < 32; ++i) {
    int c = i * 128 + tid;
    int k = c >> 2, p = c & 3;
#if USE_ASYNC_COPY
    async_g2l_b128(lds_off(&lW[k][p * 8]),
                   WtHH + (size_t)k * HDIM + n0 + p * 8);
#else
    *(U4*)&lW[k][p * 8] = *(const U4*)(WtHH + (size_t)k * HDIM + n0 + p * 8);
#endif
  }
#if USE_ASYNC_COPY
  wait_async0();
#endif
  __syncthreads();

  for (int t = 0; t < SEQLEN; ++t) {
    const unsigned short* cur = (t & 1) ? Hb : Ha;
    unsigned short*       nxt = (t & 1) ? Ha : Hb;

    // h_{t-1} bf16 (64KB) -> LDS: 4096 x 16B chunks, 32 per thread
#pragma unroll
    for (int i = 0; i < 32; ++i) {
      int c = i * 128 + tid;
      int row = c >> 7, p = c & 127;   // HDIM/8 = 128 chunks per row
#if USE_ASYNC_COPY
      async_g2l_b128(lds_off(&lH[row][p * 8]),
                     cur + (size_t)row * HDIM + p * 8);
#else
      *(U4*)&lH[row][p * 8] = *(const U4*)(cur + (size_t)row * HDIM + p * 8);
#endif
    }
#if USE_ASYNC_COPY
    wait_async0();
#endif
    __syncthreads();

    v8f acc = {};
#pragma unroll 4
    for (int kt = 0; kt < HDIM / 32; ++kt) {
      FragBF fa, fb;
      int row = mi * 16 + (lane & 15);
      int k0  = kt * 32 + ((lane >> 4) * 8);
#pragma unroll
      for (int i = 0; i < 8; ++i) {
        fa.u[i]     = lH[row][k0 + i];
        fa.u[8 + i] = lH[row][k0 + 16 + i];
      }
#pragma unroll
      for (int i = 0; i < 16; ++i) fb.u[i] = lW[k0 + (i & 7) + (i >> 3) * 16][ni * 16 + (lane & 15)];
      acc = __builtin_amdgcn_wmma_f32_16x16x32_bf16(false, fa.v, false, fb.v,
                                                    (short)0, acc, false, false);
    }

    // epilogue: h_t = tanh(xp[t] + acc); publish bf16 state (+ outputs)
    const int n = n0 + ni * 16 + (lane & 15);
#pragma unroll
    for (int r = 0; r < 8; ++r) {
      int m = mi * 16 + r + ((lane >> 4) << 3);
      size_t xi = (size_t)t * BATCH * HDIM + (size_t)m * HDIM + n;
      float v = tanhf(xp[xi] + acc[r]);
      unsigned short bv = f2bf(v);
      nxt[(size_t)m * HDIM + n] = bv;
      if (outBF) outBF[xi] = bv;
      if (outF)  outF[xi]  = v;
      if (t == SEQLEN - 1) hn[(size_t)m * HDIM + n] = v;
    }
    __syncthreads();

    // device-scope monotonic barrier across the NWG persistent blocks
    if (tid == 0) {
      __threadfence();
      atomicAdd(bar, 1u);
      const unsigned target = (unsigned)(t + 1) * NWG;
      while (atomicAdd(bar, 0u) < target) __builtin_amdgcn_s_sleep(1);
    }
    __syncthreads();
  }
}

// Seed bf16 recurrent state from h0, reset barrier counters (every launch).
__global__ void init_state(const float* __restrict__ h0,
                           unsigned short* __restrict__ H0a,
                           unsigned short* __restrict__ H1a,
                           unsigned* __restrict__ ctrs) {
  int i = blockIdx.x * blockDim.x + threadIdx.x;
  if (i < BATCH * HDIM) {
    H0a[i] = f2bf(h0[i]);
    H1a[i] = f2bf(h0[BATCH * HDIM + i]);
  }
  if (i == 0) { ctrs[0] = 0u; ctrs[1] = 0u; }
}

extern "C" void kernel_launch(void* const* d_in, const int* in_sizes, int n_in,
                              void* d_out, int out_size, void* d_ws, size_t ws_size,
                              hipStream_t stream) {
  (void)in_sizes; (void)n_in; (void)out_size; (void)ws_size;
  const float* X   = (const float*)d_in[0];   // [S,B,H]
  const float* h0  = (const float*)d_in[1];   // [L,B,H]
  const float* Wih = (const float*)d_in[2];   // [L,H,H]
  const float* Whh = (const float*)d_in[3];   // [L,H,H]
  float* out = (float*)d_out;                 // [S,B,H] ++ [L,B,H]

  const size_t SBH = (size_t)SEQLEN * BATCH * HDIM;
  const size_t HH  = (size_t)HDIM * HDIM;

  char* ws = (char*)d_ws;
  unsigned* ctrs = (unsigned*)ws;                         // 2 barrier counters
  unsigned short* Hbuf = (unsigned short*)(ws + 256);     // 4 x [B,H] bf16
  unsigned short* H0a = Hbuf;
  unsigned short* H0b = Hbuf + BATCH * HDIM;
  unsigned short* H1a = Hbuf + 2 * BATCH * HDIM;
  unsigned short* H1b = Hbuf + 3 * BATCH * HDIM;
  size_t off = 1u << 20;
  unsigned short* Xbf   = (unsigned short*)(ws + off); off += SBH * 2;  // bf16 X
  unsigned short* Ysbf  = (unsigned short*)(ws + off); off += SBH * 2;  // bf16 ys0
  unsigned short* WtIH0 = (unsigned short*)(ws + off); off += HH * 2;
  unsigned short* WtIH1 = (unsigned short*)(ws + off); off += HH * 2;
  unsigned short* WtHH0 = (unsigned short*)(ws + off); off += HH * 2;
  unsigned short* WtHH1 = (unsigned short*)(ws + off); off += HH * 2;
  float* xp0 = (float*)(ws + off);                                      // fp32 xp0

  float* outSeq = out;                        // xp1 -> ys1 (in place)
  float* hn     = out + SBH;

  const int ldsBytes = (HDIM * WPAD + BATCH * HPAD) * (int)sizeof(unsigned short);
  (void)hipFuncSetAttribute(reinterpret_cast<const void*>(rnn_scan),
                            hipFuncAttributeMaxDynamicSharedMemorySize, ldsBytes);

  // --- one-time conversions -------------------------------------------------
  init_state<<<(BATCH * HDIM + 255) / 256, 256, 0, stream>>>(h0, H0a, H1a, ctrs);
  cvt_bf16<<<(unsigned)(SBH / 8 / 256), 256, 0, stream>>>(X, Xbf, SBH / 8);
  tcvt_bf16<<<(unsigned)(HH / 256), 256, 0, stream>>>(Wih,      WtIH0);
  tcvt_bf16<<<(unsigned)(HH / 256), 256, 0, stream>>>(Wih + HH, WtIH1);
  tcvt_bf16<<<(unsigned)(HH / 256), 256, 0, stream>>>(Whh,      WtHH0);
  tcvt_bf16<<<(unsigned)(HH / 256), 256, 0, stream>>>(Whh + HH, WtHH1);

  dim3 gg(HDIM / GT_N, (SEQLEN * BATCH) / GT_M);
  // --- layer 0 --------------------------------------------------------------
  gemm_bf16<<<gg, 256, 0, stream>>>(Xbf, WtIH0, xp0);
  rnn_scan<<<NWG, 128, ldsBytes, stream>>>(xp0, nullptr, Ysbf, WtHH0,
                                           H0a, H0b, hn, &ctrs[0]);
  // --- layer 1 (xp1 lives in d_out, rewritten in place to final ys) ---------
  gemm_bf16<<<gg, 256, 0, stream>>>(Ysbf, WtIH1, outSeq);
  rnn_scan<<<NWG, 128, ldsBytes, stream>>>(outSeq, outSeq, nullptr, WtHH1,
                                           H1a, H1b, hn + BATCH * HDIM, &ctrs[1]);
}
